// VectorQuantizer_37821482008722
// MI455X (gfx1250) — matrix operand order repaired and checked by reading, not compiled, not run on hardware
//
#include <hip/hip_runtime.h>
#include <hip/hip_bf16.h>
#include <cstdint>

typedef __attribute__((ext_vector_type(16))) _Float16 v16h;
typedef __attribute__((ext_vector_type(8)))  _Float16 h8;
typedef __attribute__((ext_vector_type(8)))  float    v8f;
typedef __attribute__((ext_vector_type(4)))  float    f4;

#define NUM_EMBEDS 8192
#define EMBED_DIM  256
#define HW         1024
#define NTOK       16384      // 16 * 32 * 32 tokens
#define NUMEL      4194304    // 16 * 256 * 32 * 32

// ---------------------------------------------------------------------------
// CDNA5 async global->LDS DMA (ISA 10.7 / 15.18.3, ASYNCcnt-tracked)
// ---------------------------------------------------------------------------
__device__ __forceinline__ void async_copy_b128(const void* gaddr, unsigned ldsOff) {
  asm volatile("global_load_async_to_lds_b128 %0, %1, off"
               :: "v"(ldsOff), "v"(gaddr) : "memory");
}
__device__ __forceinline__ void wait_async_le4() {
  asm volatile("s_wait_asynccnt 0x4" ::: "memory");
}
__device__ __forceinline__ void wait_async_0() {
  asm volatile("s_wait_asynccnt 0x0" ::: "memory");
}

// ---------------------------------------------------------------------------
// Kernel 1: codebook f32 -> f16, e2[j] = ||e_j||^2, zero loss accumulator
// ---------------------------------------------------------------------------
__global__ __launch_bounds__(256) void vq_prep(const float* __restrict__ cb,
                                               _Float16* __restrict__ cbH,
                                               float* __restrict__ e2,
                                               float* __restrict__ lossAcc) {
  const int e = blockIdx.x;       // codebook row
  const int d = threadIdx.x;      // dim
  float v = cb[(size_t)e * EMBED_DIM + d];
  cbH[(size_t)e * EMBED_DIM + d] = (_Float16)v;
  float sq = v * v;
  #pragma unroll
  for (int off = 16; off > 0; off >>= 1) sq += __shfl_xor(sq, off, 32);
  __shared__ float red[8];
  const int wave = threadIdx.x >> 5, lane = threadIdx.x & 31;
  if (lane == 0) red[wave] = sq;
  __syncthreads();
  if (threadIdx.x == 0) {
    float s = 0.f;
    #pragma unroll
    for (int w = 0; w < 8; ++w) s += red[w];
    e2[e] = s;
    if (e == 0) *lossAcc = 0.f;   // deterministic re-init every launch
  }
}

// ---------------------------------------------------------------------------
// Kernel 2: fused distance-GEMM + argmin via v_wmma_f32_16x16x32_f16
//   block = 256 threads (8 wave32), owns 64 tokens, sweeps all 8192 codes.
//   B tiles double-buffered in LDS via async global->LDS DMA; two
//   independent K-split WMMA chains per wave.
//   dist proxy = e2[n] - 2 * (x . e_n)
// ---------------------------------------------------------------------------
__global__ __launch_bounds__(256) void vq_argmin(const float* __restrict__ x,
                                                 const _Float16* __restrict__ cbH,
                                                 const float* __restrict__ e2,
                                                 int* __restrict__ idxOut) {
  // A staging overlaid with the double-buffered B tiles (A needed only once)
  __shared__ union SH {
    _Float16 As[64][EMBED_DIM + 8];        // 33,792 B
    _Float16 Bs[2][32][EMBED_DIM + 8];     // 33,792 B
  } sh;
  __shared__ float redD[2][64];
  __shared__ int   redI[2][64];

  const int tid  = threadIdx.x;
  const int wave = tid >> 5;
  const int lane = tid & 31;
  const int hi   = lane >> 4;      // lane half (ISA fragment layouts)
  const int ln   = lane & 15;
  const int tt   = wave & 3;       // token subtile 0..3 (16 tokens each)
  const int cs   = wave >> 2;      // code half 0..1 (16 codes each)

  const int gtok0 = blockIdx.x * 64;          // 64 tokens per block, same batch
  const int b     = gtok0 / HW;
  const int t0    = gtok0 % HW;
  const float* xb = x + (size_t)b * EMBED_DIM * HW;

  // ---- stage A tile: x[b, k, t0+tl] -> As[tl][k] (f32 -> f16) ----
  #pragma unroll 4
  for (int it = 0; it < 64; ++it) {
    int e  = it * 256 + tid;
    int k  = e >> 6;               // dim
    int tl = e & 63;               // token within tile
    sh.As[tl][k] = (_Float16)xb[(size_t)k * HW + t0 + tl];
  }
  __syncthreads();

  // ---- build A fragments (ISA 16-bit A 16x32 layout), K = 8 chunks of 32 ----
  v16h afrag[8];
  {
    const int tokRow = tt * 16 + ln;           // lane = M row
    #pragma unroll
    for (int kc = 0; kc < 8; ++kc) {
      const h8* p0 = (const h8*)&sh.As[tokRow][kc * 32 + hi * 8];
      const h8* p1 = (const h8*)&sh.As[tokRow][kc * 32 + 16 + hi * 8];
      h8 lo = *p0, hh = *p1;
      v16h a;
      #pragma unroll
      for (int j = 0; j < 8; ++j) { a[j] = lo[j]; a[j + 8] = hh[j]; }
      afrag[kc] = a;
    }
  }
  __syncthreads();   // everyone done with As; union switches to Bs

  // ---- async copy setup: each thread moves 64 B of the 16 KB tile ----
  const int cr = tid >> 3;                    // code row 0..31
  const int ko = (tid & 7) * 32;              // 32-half chunk
  const unsigned lds0 = (unsigned)(uintptr_t)&sh.Bs[0][cr][ko];
  const unsigned lds1 = (unsigned)(uintptr_t)&sh.Bs[1][cr][ko];

  // prologue: issue tile 0 into buffer 0
  {
    const _Float16* g = cbH + (size_t)cr * EMBED_DIM + ko;
    async_copy_b128(g,      lds0);
    async_copy_b128(g +  8, lds0 + 16);
    async_copy_b128(g + 16, lds0 + 32);
    async_copy_b128(g + 24, lds0 + 48);
  }

  float bestD[8];
  int   bestI[8];
  #pragma unroll
  for (int r = 0; r < 8; ++r) { bestD[r] = 3.0e38f; bestI[r] = 0; }

  const int codeRow = cs * 16 + ln;            // B row this lane consumes

  for (int et = 0; et < NUM_EMBEDS / 32; ++et) {   // 256 code tiles of 32
    // issue next tile into the other buffer, then wait for current tile
    if (et + 1 < NUM_EMBEDS / 32) {
      const _Float16* g = cbH + (size_t)((et + 1) * 32 + cr) * EMBED_DIM + ko;
      unsigned l = ((et + 1) & 1) ? lds1 : lds0;
      async_copy_b128(g,      l);
      async_copy_b128(g +  8, l + 16);
      async_copy_b128(g + 16, l + 32);
      async_copy_b128(g + 24, l + 48);
      wait_async_le4();   // in-order retire => current tile's 4 loads done
    } else {
      wait_async_0();
    }
    __syncthreads();       // all waves' DMA for tile et visible

    const _Float16* brow = &sh.Bs[et & 1][codeRow][0];

    // ---- two independent K-split WMMA chains (K 0..127 / 128..255) ----
    v8f acc0 = {}, acc1 = {};
    #pragma unroll
    for (int kc = 0; kc < 4; ++kc) {
      const h8* q0 = (const h8*)(brow + kc * 32 + hi * 8);
      h8 b0l = q0[0], b0h = q0[2];
      v16h bf0;
      #pragma unroll
      for (int j = 0; j < 8; ++j) { bf0[j] = b0l[j]; bf0[j + 8] = b0h[j]; }
      acc0 = __builtin_amdgcn_wmma_f32_16x16x32_f16(
          false, afrag[kc], false, bf0, (short)0, acc0, false, false);

      const h8* q1 = (const h8*)(brow + (kc + 4) * 32 + hi * 8);
      h8 b1l = q1[0], b1h = q1[2];
      v16h bf1;
      #pragma unroll
      for (int j = 0; j < 8; ++j) { bf1[j] = b1l[j]; bf1[j + 8] = b1h[j]; }
      acc1 = __builtin_amdgcn_wmma_f32_16x16x32_f16(
          false, afrag[kc + 4], false, bf1, (short)0, acc1, false, false);
    }

    // ---- argmin update: lane holds column n for rows (hi*8 + r) ----
    const int  code = et * 32 + cs * 16 + ln;
    const float e2v = e2[code];
    #pragma unroll
    for (int r = 0; r < 8; ++r) {
      float s  = acc0[r] + acc1[r];
      float dd = __builtin_fmaf(-2.0f, s, e2v);
      bool better = dd < bestD[r];                // strict <: first occurrence
      bestD[r] = better ? dd : bestD[r];
      bestI[r] = better ? code : bestI[r];
    }
    __syncthreads();   // all reads of buffer (et&1) done before its reuse
  }

  // ---- reduce across the 16 lanes sharing each row (columns n) ----
  #pragma unroll
  for (int r = 0; r < 8; ++r) {
    float d = bestD[r]; int i = bestI[r];
    #pragma unroll
    for (int off = 1; off < 16; off <<= 1) {
      float od = __shfl_xor(d, off, 32);
      int   oi = __shfl_xor(i, off, 32);
      if (od < d || (od == d && oi < i)) { d = od; i = oi; }
    }
    bestD[r] = d; bestI[r] = i;
  }
  if (ln == 0) {   // lanes 0 (rows 0..7) and 16 (rows 8..15) publish
    #pragma unroll
    for (int r = 0; r < 8; ++r) {
      int row = hi * 8 + r;
      redD[cs][tt * 16 + row] = bestD[r];
      redI[cs][tt * 16 + row] = bestI[r];
    }
  }
  __syncthreads();

  // ---- reduce across the 2 code halves, emit final index per token ----
  if (tid < 64) {
    float d0 = redD[0][tid], d1 = redD[1][tid];
    int   i0 = redI[0][tid], i1 = redI[1][tid];
    bool take1 = (d1 < d0) || (d1 == d0 && i1 < i0);
    idxOut[gtok0 + tid] = take1 ? i1 : i0;
  }
}

// ---------------------------------------------------------------------------
// Kernel 3: gather quantized output via LDS transpose (fully coalesced) +
//           loss partial sums. Block = 16 tokens x 256 dims.
// ---------------------------------------------------------------------------
__global__ __launch_bounds__(256) void vq_output(const float* __restrict__ x,
                                                 const float* __restrict__ cb,
                                                 const int* __restrict__ idx,
                                                 float* __restrict__ out,
                                                 float* __restrict__ lossAcc) {
  __shared__ int   sIdx[16];
  __shared__ float Q[16][EMBED_DIM + 4];    // +4 pad: 16B-aligned rows, no bank conflict

  const int tid   = threadIdx.x;
  const int gtok0 = blockIdx.x * 16;
  const int b     = gtok0 / HW;
  const int t0    = gtok0 % HW;

  if (tid < 16) sIdx[tid] = idx[gtok0 + tid];
  __syncthreads();

  // phase 1: coalesced read of 16 codebook rows -> LDS
  {
    const int i  = tid >> 4;                 // token 0..15
    const int d0 = (tid & 15) * 16;          // 16-float chunk
    const f4* s4 = (const f4*)(cb + (size_t)sIdx[i] * EMBED_DIM + d0);
    f4 a0 = s4[0], a1 = s4[1], a2 = s4[2], a3 = s4[3];
    f4* d4 = (f4*)&Q[i][d0];
    d4[0] = a0; d4[1] = a1; d4[2] = a2; d4[3] = a3;
  }
  __syncthreads();

  // phase 2: transpose out of LDS, write [b, d, t0+tl] in 64B segments
  float sq = 0.f;
  #pragma unroll
  for (int it = 0; it < 16; ++it) {
    const int d  = it * 16 + (tid >> 4);
    const int tl = tid & 15;
    const float q = Q[tl][d];
    const size_t o = (size_t)b * (EMBED_DIM * HW) + (size_t)d * HW + t0 + tl;
    const float xv = x[o];
    out[o] = q;                              // quantized_st == quantized numerically
    const float df = q - xv;
    sq = __builtin_fmaf(df, df, sq);
  }

  #pragma unroll
  for (int off = 16; off > 0; off >>= 1) sq += __shfl_xor(sq, off, 32);
  __shared__ float red[8];
  const int wave = tid >> 5, lane = tid & 31;
  if (lane == 0) red[wave] = sq;
  __syncthreads();
  if (tid == 0) {
    float s = 0.f;
    #pragma unroll
    for (int w = 0; w < 8; ++w) s += red[w];
    atomicAdd(lossAcc, s);
  }
}

// ---------------------------------------------------------------------------
// Kernel 4: vq_loss = (BETA + 1) * mean((q - x)^2)
// ---------------------------------------------------------------------------
__global__ void vq_finalize(const float* __restrict__ lossAcc,
                            float* __restrict__ lossOut) {
  *lossOut = (*lossAcc) * 1.25f / (float)NUMEL;
}

// ---------------------------------------------------------------------------
extern "C" void kernel_launch(void* const* d_in, const int* in_sizes, int n_in,
                              void* d_out, int out_size, void* d_ws, size_t ws_size,
                              hipStream_t stream) {
  const float* x  = (const float*)d_in[0];   // [16, 256, 32, 32] f32
  const float* cb = (const float*)d_in[1];   // [8192, 256] f32

  char* ws = (char*)d_ws;
  _Float16* cbH  = (_Float16*)ws;                                  // 4,194,304 B
  float*    e2   = (float*)(ws + 4194304);                         //    32,768 B
  int*      idx  = (int*)  (ws + 4194304 + 32768);                 //    65,536 B
  float*    acc  = (float*)(ws + 4194304 + 32768 + 65536);         //         4 B

  float* out     = (float*)d_out;            // quantized_st [NUMEL] then loss [1]
  float* lossOut = out + NUMEL;

  vq_prep    <<<NUM_EMBEDS,  256, 0, stream>>>(cb, cbH, e2, acc);
  vq_argmin  <<<NTOK / 64,   256, 0, stream>>>(x, cbH, e2, idx);
  vq_output  <<<NTOK / 16,   256, 0, stream>>>(x, cb, idx, out, acc);
  vq_finalize<<<1, 1, 0, stream>>>(acc, lossOut);
}
